// MultiHeadAttention_70720931496315
// MI455X (gfx1250) — matrix operand order repaired and checked by reading, not compiled, not run on hardware
//
#include <hip/hip_runtime.h>
#include <hip/hip_bf16.h>

// MHA forward for B=2, S=2048, D=1024, H=16, HD=64 (fp32 in/out).
// Compute path: f16 WMMA (v_wmma_f32_16x16x32_f16) with f32 accumulation,
// f32 online softmax. Pipeline: cvt -> QKV GEMM -> flash attn -> out GEMM.
// Round 2: batch all fragment loads of a k-step into live registers and
// fence with sched_barrier so loads clause up and overlap the XDL pipe.

typedef _Float16 half_t;
typedef __attribute__((ext_vector_type(16))) _Float16 v16h;
typedef __attribute__((ext_vector_type(8)))  _Float16 v8h;
typedef __attribute__((ext_vector_type(8)))  float    v8f;

#define B_  2
#define S_  2048
#define D_  1024
#define H_  16
#define HD_ 64

__device__ __forceinline__ v16h make16(v8h lo, v8h hi) {
  union { v16h v; v8h h[2]; } u;
  u.h[0] = lo; u.h[1] = hi;
  return u.v;
}

#define WMMA_F16(a, b, c) \
  __builtin_amdgcn_wmma_f32_16x16x32_f16(false, (a), false, (b), (short)0, (c), false, false)

// ---------------------------------------------------------------------------
// Kernel 1: fp32 -> fp16 elementwise convert (4 elements / thread)
// ---------------------------------------------------------------------------
__global__ __launch_bounds__(256) void cvt_f32_f16_kernel(
    const float* __restrict__ src, half_t* __restrict__ dst, int n4) {
  int i = blockIdx.x * blockDim.x + threadIdx.x;
  if (i >= n4) return;
  const float4 v = ((const float4*)src)[i];
  union { half_t h[4]; uint2 u; } o;
  o.h[0] = (half_t)v.x; o.h[1] = (half_t)v.y;
  o.h[2] = (half_t)v.z; o.h[3] = (half_t)v.w;
  ((uint2*)dst)[i] = o.u;
}

// ---------------------------------------------------------------------------
// Kernel 2: transpose + convert: src fp32 [R,C] -> dst f16 [C,R]
// (weights become [N,K] so WMMA B-fragments load K-contiguous)
// ---------------------------------------------------------------------------
__global__ __launch_bounds__(256) void transpose_cvt_kernel(
    const float* __restrict__ src, half_t* __restrict__ dst, int R, int C) {
  __shared__ float tile[32][33];
  const int c0 = blockIdx.x * 32, r0 = blockIdx.y * 32;
  const int tx = threadIdx.x, ty = threadIdx.y;  // block (32,8)
  #pragma unroll
  for (int i = 0; i < 32; i += 8)
    tile[ty + i][tx] = src[(size_t)(r0 + ty + i) * C + c0 + tx];
  __syncthreads();
  #pragma unroll
  for (int i = 0; i < 32; i += 8)
    dst[(size_t)(c0 + ty + i) * R + r0 + tx] = (half_t)tile[tx][ty + i];
}

// ---------------------------------------------------------------------------
// Wave-level GEMM tile: acc[8] (16 rows x 128 cols) = A[16,K] x Bt[128,K]^T
// A row-major [M,K] f16, Bt row-major [N,K] f16 (i.e. B column-major).
// All 9 fragments of a k-step are loaded into live registers first (one
// clause, one wait), then 8 WMMAs issue back-to-back; next k-step's loads
// overlap the XDL pipe (only WAR hazards on the fragment registers).
// ---------------------------------------------------------------------------
__device__ __forceinline__ void gemm_tile_wave(
    const half_t* __restrict__ A, const half_t* __restrict__ Bt, int K,
    int rowBase, int colBase, int m16, int hh, v8f acc[8]) {
  const half_t* arow = A + (size_t)(rowBase + m16) * K;
  for (int k0 = 0; k0 < K; k0 += 32) {
    // A fragment 16x32: halfs[0..15]=K(16*hh .. 16*hh+15)
    const v16h a = make16(*(const v8h*)(arow + k0 + 16 * hh),
                          *(const v8h*)(arow + k0 + 16 * hh + 8));
    v16h bb[8];
    #pragma unroll
    for (int j = 0; j < 8; ++j) {
      // B fragment 32x16: lane col = m16, halfs = K(16*hh .. 16*hh+15)
      const half_t* br = Bt + (size_t)(colBase + j * 16 + m16) * K + k0 + 16 * hh;
      bb[j] = make16(*(const v8h*)br, *(const v8h*)(br + 8));
    }
    __builtin_amdgcn_sched_barrier(0);   // keep loads clustered before WMMAs
    #pragma unroll
    for (int j = 0; j < 8; ++j)
      acc[j] = WMMA_F16(a, bb[j], acc[j]);
    __builtin_amdgcn_sched_barrier(0);
  }
}

// ---------------------------------------------------------------------------
// Kernel 3: QKV projection. xh[4096,1024] x WqkvT[3072,1024]^T + bqkv.
// Scatter: Q,K -> [B,H,S,HD] f16 ; V -> transposed [B,H,HD,S] f16.
// ---------------------------------------------------------------------------
__global__ __launch_bounds__(256) void gemm_qkv_kernel(
    const half_t* __restrict__ xh, const half_t* __restrict__ WqkvT,
    const float* __restrict__ bqkv,
    half_t* __restrict__ Qb, half_t* __restrict__ Kb, half_t* __restrict__ Vt) {
  const int lane = threadIdx.x & 31, w = threadIdx.x >> 5;
  const int m16 = lane & 15, hh = lane >> 4;
  const int rowBase = blockIdx.y * 128 + w * 16;
  const int colBase = blockIdx.x * 128;
  v8f acc[8] = {};
  gemm_tile_wave(xh, WqkvT, D_, rowBase, colBase, m16, hh, acc);
  #pragma unroll
  for (int j = 0; j < 8; ++j) {
    const int n = colBase + j * 16 + m16;       // [0, 3072)
    const float bias = bqkv[n];
    const int which = n >> 10;                  // 0=q 1=k 2=v
    const int d = n & 1023;
    const int h = d >> 6, hd = d & 63;
    #pragma unroll
    for (int v = 0; v < 8; ++v) {
      const int m = rowBase + v + 8 * hh;       // [0, 4096)
      const int b = m >> 11, s = m & 2047;
      const float val = acc[j][v] + bias;
      const size_t bh = (size_t)(b * H_ + h);
      if (which == 0)      Qb[(bh * S_ + s) * HD_ + hd] = (half_t)val;
      else if (which == 1) Kb[(bh * S_ + s) * HD_ + hd] = (half_t)val;
      else                 Vt[(bh * HD_ + hd) * S_ + s] = (half_t)val;
    }
  }
}

// ---------------------------------------------------------------------------
// Kernel 4: flash attention. One block = (b,h) x 128 query rows; 8 waves,
// wave owns 16 q rows. Streams 64-key tiles with causal mask + log(mask),
// f32 online softmax, P relayout (C->A) through LDS, PV via WMMA.
// ---------------------------------------------------------------------------
__global__ __launch_bounds__(256) void flash_attn_kernel(
    const half_t* __restrict__ Qb, const half_t* __restrict__ Kb,
    const half_t* __restrict__ Vt, const float* __restrict__ mask,
    half_t* __restrict__ attnh) {
  __shared__ _Float16 lds[8 * 16 * 64];        // 2 KB per wave
  const int lane = threadIdx.x & 31;
  const int w    = threadIdx.x >> 5;
  const int m16  = lane & 15;
  const int hh   = lane >> 4;
  const int bh   = blockIdx.x;
  const int b    = bh >> 4;                    // / H_
  const int h    = bh & 15;
  const int s0w  = blockIdx.y * 128 + w * 16;  // wave's first q row

  // Q A-fragments (16 x 64), kept in registers for the whole kernel.
  const half_t* qrow = Qb + ((size_t)bh * S_ + (s0w + m16)) * HD_;
  v16h qa[2];
  #pragma unroll
  for (int kb = 0; kb < 2; ++kb)
    qa[kb] = make16(*(const v8h*)(qrow + kb * 32 + 16 * hh),
                    *(const v8h*)(qrow + kb * 32 + 16 * hh + 8));

  v8f of[4] = {};
  float mrun[8], lrun[8];
  #pragma unroll
  for (int v = 0; v < 8; ++v) { mrun[v] = -1.0e30f; lrun[v] = 0.0f; }

  const float* mrow = mask + (size_t)b * S_;
  _Float16* pbase = lds + w * (16 * 64);
  const int ntiles = (blockIdx.y + 1) * 2;     // keys up to block diagonal

  for (int t = 0; t < ntiles; ++t) {
    const int z0 = t * 64;
    // ---- load all 8 K B-fragments, then 8 score WMMAs (4 indep chains) ----
    v16h bk[8];
    #pragma unroll
    for (int zf = 0; zf < 4; ++zf) {
      const half_t* krow = Kb + ((size_t)bh * S_ + (z0 + zf * 16 + m16)) * HD_;
      #pragma unroll
      for (int kb = 0; kb < 2; ++kb)
        bk[zf * 2 + kb] = make16(*(const v8h*)(krow + kb * 32 + 16 * hh),
                                 *(const v8h*)(krow + kb * 32 + 16 * hh + 8));
    }
    __builtin_amdgcn_sched_barrier(0);
    v8f sf[4] = {};
    #pragma unroll
    for (int zf = 0; zf < 4; ++zf)
      #pragma unroll
      for (int kb = 0; kb < 2; ++kb)
        sf[zf] = WMMA_F16(qa[kb], bk[zf * 2 + kb], sf[zf]);
    __builtin_amdgcn_sched_barrier(0);
    // ---- scale + log(mask) + causal ----
    #pragma unroll
    for (int zf = 0; zf < 4; ++zf) {
      const int z = z0 + zf * 16 + m16;
      const float lm = fmaxf(__logf(mrow[z]), -1.0e30f);
      #pragma unroll
      for (int v = 0; v < 8; ++v) {
        const int qr = s0w + v + 8 * hh;
        const float x = sf[zf][v] * 0.125f + lm;   // 1/sqrt(64)
        sf[zf][v] = (z > qr) ? -1.0e30f : x;
      }
    }
    // ---- online softmax per row (row v+8*hh lives in 16-lane half) ----
    #pragma unroll
    for (int v = 0; v < 8; ++v) {
      float t0 = fmaxf(fmaxf(sf[0][v], sf[1][v]), fmaxf(sf[2][v], sf[3][v]));
      t0 = fmaxf(t0, __shfl_xor(t0, 1, 32));
      t0 = fmaxf(t0, __shfl_xor(t0, 2, 32));
      t0 = fmaxf(t0, __shfl_xor(t0, 4, 32));
      t0 = fmaxf(t0, __shfl_xor(t0, 8, 32));
      const float mn   = fmaxf(mrun[v], t0);
      const float corr = __expf(mrun[v] - mn);
      mrun[v] = mn;
      lrun[v] *= corr;
      #pragma unroll
      for (int f = 0; f < 4; ++f) of[f][v] *= corr;
      float rs = 0.0f;
      #pragma unroll
      for (int zf = 0; zf < 4; ++zf) {
        const float p = __expf(sf[zf][v] - mn);
        sf[zf][v] = p;
        rs += p;
      }
      rs += __shfl_xor(rs, 1, 32);
      rs += __shfl_xor(rs, 2, 32);
      rs += __shfl_xor(rs, 4, 32);
      rs += __shfl_xor(rs, 8, 32);
      lrun[v] += rs;
    }
    // ---- P: C-layout f32 -> A-layout f16 via per-wave LDS region ----
    #pragma unroll
    for (int zf = 0; zf < 4; ++zf)
      #pragma unroll
      for (int v = 0; v < 8; ++v)
        pbase[(v + 8 * hh) * 64 + zf * 16 + m16] = (half_t)sf[zf][v];
    asm volatile("s_wait_dscnt 0x0" ::: "memory");  // DS in-order per wave
    const _Float16* prow = pbase + m16 * 64;
    v16h pa[2];
    #pragma unroll
    for (int kb = 0; kb < 2; ++kb)
      pa[kb] = make16(*(const v8h*)(prow + kb * 32 + 16 * hh),
                      *(const v8h*)(prow + kb * 32 + 16 * hh + 8));
    // ---- load all 8 V B-fragments, then 8 PV WMMAs (4 indep chains) ----
    v16h vb[8];
    #pragma unroll
    for (int df = 0; df < 4; ++df) {
      const half_t* vr = Vt + ((size_t)bh * HD_ + df * 16 + m16) * S_ + z0;
      #pragma unroll
      for (int kb = 0; kb < 2; ++kb)
        vb[df * 2 + kb] = make16(*(const v8h*)(vr + kb * 32 + 16 * hh),
                                 *(const v8h*)(vr + kb * 32 + 16 * hh + 8));
    }
    __builtin_amdgcn_sched_barrier(0);
    #pragma unroll
    for (int df = 0; df < 4; ++df)
      #pragma unroll
      for (int kb = 0; kb < 2; ++kb)
        of[df] = WMMA_F16(pa[kb], vb[df * 2 + kb], of[df]);
    __builtin_amdgcn_sched_barrier(0);
  }
  // ---- normalize + write attn out [B,S,D] f16 ----
  #pragma unroll
  for (int v = 0; v < 8; ++v) {
    const float inv = 1.0f / lrun[v];
    const int s = s0w + v + 8 * hh;
    half_t* orow = attnh + ((size_t)b * S_ + s) * D_ + h * HD_;
    #pragma unroll
    for (int df = 0; df < 4; ++df)
      orow[df * 16 + m16] = (half_t)(of[df][v] * inv);
  }
}

// ---------------------------------------------------------------------------
// Kernel 5: output projection. attnh[4096,1024] x WoT[1024,1024]^T + bo -> f32
// ---------------------------------------------------------------------------
__global__ __launch_bounds__(256) void gemm_out_kernel(
    const half_t* __restrict__ attnh, const half_t* __restrict__ WoT,
    const float* __restrict__ bo, float* __restrict__ out) {
  const int lane = threadIdx.x & 31, w = threadIdx.x >> 5;
  const int m16 = lane & 15, hh = lane >> 4;
  const int rowBase = blockIdx.y * 128 + w * 16;
  const int colBase = blockIdx.x * 128;
  v8f acc[8] = {};
  gemm_tile_wave(attnh, WoT, D_, rowBase, colBase, m16, hh, acc);
  #pragma unroll
  for (int j = 0; j < 8; ++j) {
    const int n = colBase + j * 16 + m16;
    const float bias = bo[n];
    #pragma unroll
    for (int v = 0; v < 8; ++v) {
      const int m = rowBase + v + 8 * hh;
      out[(size_t)m * D_ + n] = acc[j][v] + bias;
    }
  }
}

// ---------------------------------------------------------------------------
extern "C" void kernel_launch(void* const* d_in, const int* in_sizes, int n_in,
                              void* d_out, int out_size, void* d_ws, size_t ws_size,
                              hipStream_t stream) {
  const float* x    = (const float*)d_in[0];
  const float* mask = (const float*)d_in[1];
  const float* Wqkv = (const float*)d_in[2];
  const float* bqkv = (const float*)d_in[3];
  const float* Wo   = (const float*)d_in[4];
  const float* bo   = (const float*)d_in[5];
  float* out = (float*)d_out;
  (void)in_sizes; (void)n_in; (void)out_size; (void)ws_size;

  char* ws = (char*)d_ws;
  size_t off = 0;
  auto alloc = [&](size_t bytes) {
    char* p = ws + off;
    off += (bytes + 255) & ~(size_t)255;
    return p;
  };
  half_t* xh    = (half_t*)alloc((size_t)B_ * S_ * D_ * 2);        // 8 MB
  half_t* WqkvT = (half_t*)alloc((size_t)3 * D_ * D_ * 2);         // 6 MB
  half_t* WoT   = (half_t*)alloc((size_t)D_ * D_ * 2);             // 2 MB
  half_t* Qb    = (half_t*)alloc((size_t)B_ * H_ * S_ * HD_ * 2);  // 8 MB
  half_t* Kb    = (half_t*)alloc((size_t)B_ * H_ * S_ * HD_ * 2);  // 8 MB
  half_t* Vt    = (half_t*)alloc((size_t)B_ * H_ * S_ * HD_ * 2);  // 8 MB
  half_t* attnh = (half_t*)alloc((size_t)B_ * S_ * D_ * 2);        // 8 MB

  const int n4 = B_ * S_ * D_ / 4;
  cvt_f32_f16_kernel<<<(n4 + 255) / 256, 256, 0, stream>>>(x, xh, n4);
  transpose_cvt_kernel<<<dim3(3 * D_ / 32, D_ / 32), dim3(32, 8), 0, stream>>>(
      Wqkv, WqkvT, D_, 3 * D_);
  transpose_cvt_kernel<<<dim3(D_ / 32, D_ / 32), dim3(32, 8), 0, stream>>>(
      Wo, WoT, D_, D_);
  gemm_qkv_kernel<<<dim3(3 * D_ / 128, B_ * S_ / 128), 256, 0, stream>>>(
      xh, WqkvT, bqkv, Qb, Kb, Vt);
  flash_attn_kernel<<<dim3(B_ * H_, S_ / 128), 256, 0, stream>>>(
      Qb, Kb, Vt, mask, attnh);
  gemm_out_kernel<<<dim3(D_ / 128, B_ * S_ / 128), 256, 0, stream>>>(
      attnh, WoT, bo, out);
}
